// DeformLayer_43808666419831
// MI455X (gfx1250) — matrix operand order repaired and checked by reading, not compiled, not run on hardware
//
#include <hip/hip_runtime.h>
#include <hip/hip_bf16.h>

typedef __attribute__((ext_vector_type(16))) __bf16 v16bf;
typedef __attribute__((ext_vector_type(8)))  float  v8f;

#define Hh   128
#define Ww   128
#define Cc   64
#define Co   64
#define Bb   8
#define KT   9             // 3x3 taps
#define Kdim (KT * Cc)     // 576
#define NCH  18            // K chunks of 32
#define HW   (Hh * Ww)

#define WB_WORDS   (NCH * 4 * 32 * 8)   // 18432 u32 = 73728 B (main weights)
#define WOFF_WORDS (NCH * 2 * 32 * 8)   // 9216 u32  = 36864 B (offset-conv weights)

union AFrag { v16bf v; __bf16 e[16]; unsigned u[8]; };
union BFrag { v16bf v; uint4 q[2]; };

static __device__ __forceinline__ unsigned short f2bf(float f) {
    unsigned u = __float_as_uint(f);
    unsigned r = (u + 0x7FFFu + ((u >> 16) & 1u)) >> 16;   // RNE
    return (unsigned short)r;
}
static __device__ __forceinline__ unsigned packbf(float a, float b) {
    return (unsigned)f2bf(a) | ((unsigned)f2bf(b) << 16);
}

// Async-copy n128*256 b128 chunks from global to LDS, 256 threads.
static __device__ __forceinline__ void async_stage(const unsigned* g, void* lds, int n128) {
    unsigned lds_base = (unsigned)(unsigned long long)lds;
    unsigned long long gbase = (unsigned long long)g;
    for (int i = 0; i < n128; ++i) {
        unsigned byteoff = ((unsigned)threadIdx.x + (unsigned)i * 256u) * 16u;
        unsigned ldsaddr = lds_base + byteoff;
        asm volatile("global_load_async_to_lds_b128 %0, %1, %2"
                     :: "v"(ldsaddr), "v"(byteoff), "s"(gbase)
                     : "memory");
    }
    asm volatile("s_wait_asynccnt 0" ::: "memory");
}

// ---------------------------------------------------------------------------
// Kernel 0: pack both weight tensors to bf16 in per-lane WMMA B-fragment
// order: word index = ((chunk*T + t)*32 + lane)*8 + v, where the lane holds
// K = chunk*32 + 16*(lane>=16) + 2v (+1 in high half), N = t*16 + lane%16.
// K is reordered as k = kk*64 + c (tap-major, channel-minor).
// ---------------------------------------------------------------------------
__global__ __launch_bounds__(256) void dcn_prep(const float* __restrict__ w_dcn,
                                                const float* __restrict__ w_off,
                                                unsigned* __restrict__ wB,
                                                unsigned* __restrict__ wOffB) {
    int i = blockIdx.x * 256 + threadIdx.x;
    if (i < WB_WORDS) {                         // main GEMM weights, 4 N-tiles
        int v = i & 7, lane = (i >> 3) & 31, t = (i >> 8) & 3, ch = i >> 10;
        int k = ch * 32 + (lane >> 4) * 16 + 2 * v;
        int n = t * 16 + (lane & 15);
        int kk = k >> 6, c = k & 63;
        float f0 = w_dcn[(n * Cc + c) * 9 + kk];
        float f1 = w_dcn[(n * Cc + c + 1) * 9 + kk];
        wB[i] = packbf(f0, f1);
    } else if (i < WB_WORDS + WOFF_WORDS) {     // offset-conv weights, 2 N-tiles
        int j = i - WB_WORDS;
        int v = j & 7, lane = (j >> 3) & 31, t = (j >> 8) & 1, ch = j >> 9;
        int k = ch * 32 + (lane >> 4) * 16 + 2 * v;
        int n = t * 16 + (lane & 15);
        int kk = k >> 6, c = k & 63;
        float f0 = (n < 27) ? w_off[(n * Cc + c) * 9 + kk] : 0.0f;
        float f1 = (n < 27) ? w_off[(n * Cc + c + 1) * 9 + kk] : 0.0f;
        wOffB[j] = packbf(f0, f1);
    }
}

// ---------------------------------------------------------------------------
// Kernel 1: offset conv (27 ch, padded to 32) as implicit GEMM on WMMA.
// Block = one (b,h) row, 8 waves x 16 pixels. Each wave: 16x32 tile,
// 18 K-chunks -> 36 WMMAs. Zero-padding via clamped address + 0/1 multiply
// (no EXEC divergence). Finalize via LDS transpose of the 16x27 tile.
// ---------------------------------------------------------------------------
__global__ __launch_bounds__(256) void dcn_offsets(const float* __restrict__ x,
                                                   const unsigned* __restrict__ wOffB,
                                                   const float* __restrict__ b_off,
                                                   float* __restrict__ pyA,
                                                   float* __restrict__ pxA,
                                                   float* __restrict__ mA) {
    __shared__ uint4 lds_w[WOFF_WORDS / 4];     // 36864 B, fragment-swizzled
    __shared__ float lds_om[8][16][32];         // 16384 B: per-wave 16 pixels x 32 ch

    async_stage(wOffB, lds_w, WOFF_WORDS / 1024);
    __syncthreads();

    int tid  = threadIdx.x;
    int wave = tid >> 5;
    int lane = tid & 31;
    int ml   = lane & 15;
    int hi   = lane >> 4;

    int b  = blockIdx.x / Hh;
    int h  = blockIdx.x % Hh;
    int w0 = wave * 16;
    int wpix = w0 + ml;

    const float* xb = x + (size_t)b * Cc * HW;

    v8f acc[2] = {};
    for (int kk = 0; kk < 9; ++kk) {
        int iy  = h - 1 + kk / 3;
        int ixl = wpix - 1 + kk % 3;
        bool vv = ((unsigned)iy < (unsigned)Hh) && ((unsigned)ixl < (unsigned)Ww);
        float vmul = vv ? 1.0f : 0.0f;
        int iyc = min(max(iy, 0), Hh - 1);
        int ixc = min(max(ixl, 0), Ww - 1);
        const float* xp = xb + iyc * Ww + ixc;  // always-safe address

        #pragma unroll
        for (int half = 0; half < 64; half += 32) {
            AFrag Af;
            int cb = half + (hi ? 16 : 0);
            #pragma unroll
            for (int v = 0; v < 8; ++v) {
                int c0 = cb + 2 * v;
                float s0 = xp[c0 * HW] * vmul;
                float s1 = xp[c0 * HW + HW] * vmul;
                Af.e[2 * v]     = (__bf16)s0;
                Af.e[2 * v + 1] = (__bf16)s1;
            }
            int ch = kk * 2 + (half >> 5);
            #pragma unroll
            for (int t = 0; t < 2; ++t) {
                BFrag Bf;
                int fb = ((ch * 2 + t) * 32 + lane) * 2;
                Bf.q[0] = lds_w[fb];
                Bf.q[1] = lds_w[fb + 1];
                acc[t] = __builtin_amdgcn_wmma_f32_16x16x32_bf16(
                    false, Af.v, false, Bf.v, (short)0, acc[t], false, false);
            }
        }
    }

    // D layout: VGPR r -> M = r + 8*hi, N = ml. Add bias, park in LDS.
    #pragma unroll
    for (int t = 0; t < 2; ++t) {
        int n = t * 16 + ml;
        float bias = (n < 27) ? b_off[n] : 0.0f;
        #pragma unroll
        for (int r = 0; r < 8; ++r)
            lds_om[wave][r + hi * 8][n] = acc[t][r] + bias;
    }
    __syncthreads();

    if (lane < 16) {                            // one pixel per lane
        int w = w0 + lane;
        const float* om = &lds_om[wave][lane][0];
        int sb = (b * KT * Hh + h) * Ww + w;
        #pragma unroll
        for (int k = 0; k < 9; ++k) {
            float dy = om[2 * k];
            float dx = om[2 * k + 1];
            float ms = 1.0f / (1.0f + __expf(-om[18 + k]));
            int idx = sb + k * HW;
            pyA[idx] = dy + (float)(h - 1 + k / 3);
            pxA[idx] = dx + (float)(w - 1 + k % 3);
            mA[idx]  = ms;
        }
    }
}

// ---------------------------------------------------------------------------
// Kernel 2: deformable gather + implicit GEMM via v_wmma_f32_16x16x32_bf16.
// One block per (b,h) row: 8 waves x 16 w-positions. Each wave: 16(M) x 64(N),
// K = 576 in 18 chunks -> 72 WMMAs. B fragments: two ds_load_b128 each.
// ---------------------------------------------------------------------------
__global__ __launch_bounds__(256) void dcn_main(const float* __restrict__ x,
                                                const unsigned* __restrict__ wB,
                                                const float* __restrict__ pyA,
                                                const float* __restrict__ pxA,
                                                const float* __restrict__ mA,
                                                float* __restrict__ out) {
    __shared__ uint4 lds_w[WB_WORDS / 4];       // 73728 B, fragment-swizzled

    async_stage(wB, lds_w, WB_WORDS / 1024);
    __syncthreads();

    int tid  = threadIdx.x;
    int wave = tid >> 5;
    int lane = tid & 31;
    int ml   = lane & 15;
    int hi   = lane >> 4;

    int b  = blockIdx.x / Hh;
    int h  = blockIdx.x % Hh;
    int w0 = wave * 16;
    int mw = w0 + ml;

    const float* xb = x + (size_t)b * Cc * HW;

    v8f acc[4] = {};

    for (int kk = 0; kk < 9; ++kk) {
        int sbase = ((b * KT + kk) * Hh + h) * Ww + mw;
        __builtin_prefetch(&pyA[sbase + HW], 0, 0);   // global_prefetch_b8

        float py = pyA[sbase], px = pxA[sbase], msk = mA[sbase];
        float y0f = floorf(py), x0f = floorf(px);
        int   y0 = (int)y0f,   x0 = (int)x0f;
        float fy = py - y0f,   fx = px - x0f;
        int   y1 = y0 + 1,     x1 = x0 + 1;

        float wt[4];
        wt[0] = (1.0f - fy) * (1.0f - fx);
        wt[1] = (1.0f - fy) * fx;
        wt[2] = fy * (1.0f - fx);
        wt[3] = fy * fx;
        bool vy0 = (unsigned)y0 < (unsigned)Hh, vy1 = (unsigned)y1 < (unsigned)Hh;
        bool vx0 = (unsigned)x0 < (unsigned)Ww, vx1 = (unsigned)x1 < (unsigned)Ww;
        wt[0] = (vy0 && vx0) ? wt[0] * msk : 0.0f;
        wt[1] = (vy0 && vx1) ? wt[1] * msk : 0.0f;
        wt[2] = (vy1 && vx0) ? wt[2] * msk : 0.0f;
        wt[3] = (vy1 && vx1) ? wt[3] * msk : 0.0f;

        int cy0 = min(max(y0, 0), Hh - 1), cy1 = min(max(y1, 0), Hh - 1);
        int cx0 = min(max(x0, 0), Ww - 1), cx1 = min(max(x1, 0), Ww - 1);
        int idx[4];
        idx[0] = cy0 * Ww + cx0; idx[1] = cy0 * Ww + cx1;
        idx[2] = cy1 * Ww + cx0; idx[3] = cy1 * Ww + cx1;

        #pragma unroll
        for (int half = 0; half < 64; half += 32) {
            // ---- A fragment (16-bit A 16x32 layout), bilinear-sampled bf16 ----
            AFrag Af;
            int cb = half + (hi ? 16 : 0);
            #pragma unroll
            for (int v = 0; v < 8; ++v) {
                int c0 = cb + 2 * v;
                const float* p0 = xb + c0 * HW;
                const float* p1 = p0 + HW;
                float s0 = fmaf(wt[0], p0[idx[0]],
                           fmaf(wt[1], p0[idx[1]],
                           fmaf(wt[2], p0[idx[2]], wt[3] * p0[idx[3]])));
                float s1 = fmaf(wt[0], p1[idx[0]],
                           fmaf(wt[1], p1[idx[1]],
                           fmaf(wt[2], p1[idx[2]], wt[3] * p1[idx[3]])));
                Af.e[2 * v]     = (__bf16)s0;
                Af.e[2 * v + 1] = (__bf16)s1;
            }

            // ---- B fragments: two aligned ds_load_b128 each, then WMMA ----
            int ch = kk * 2 + (half >> 5);
            #pragma unroll
            for (int t = 0; t < 4; ++t) {
                BFrag Bf;
                int fb = ((ch * 4 + t) * 32 + lane) * 2;
                Bf.q[0] = lds_w[fb];
                Bf.q[1] = lds_w[fb + 1];
                acc[t] = __builtin_amdgcn_wmma_f32_16x16x32_bf16(
                    false, Af.v, false, Bf.v, (short)0, acc[t], false, false);
            }
        }
    }

    // ---- store D: VGPR r -> M = r + 8*hi, N = ml ----
    #pragma unroll
    for (int t = 0; t < 4; ++t) {
        int o = t * 16 + ml;
        float* ob = out + (((size_t)b * Co + o) * Hh + h) * Ww + w0;
        #pragma unroll
        for (int r = 0; r < 8; ++r) {
            ob[r + hi * 8] = acc[t][r];
        }
    }
}

// ---------------------------------------------------------------------------
extern "C" void kernel_launch(void* const* d_in, const int* in_sizes, int n_in,
                              void* d_out, int out_size, void* d_ws, size_t ws_size,
                              hipStream_t stream) {
    const float* x     = (const float*)d_in[0];
    const float* w_off = (const float*)d_in[1];
    const float* b_off = (const float*)d_in[2];
    const float* w_dcn = (const float*)d_in[3];
    float* out = (float*)d_out;

    char* ws = (char*)d_ws;
    unsigned* wB    = (unsigned*)ws;                               // 73728 B
    unsigned* wOffB = wB + WB_WORDS;                               // 36864 B
    float* pyA = (float*)(ws + (WB_WORDS + WOFF_WORDS) * 4);       // 4.5 MB each
    float* pxA = pyA + Bb * KT * HW;
    float* mA  = pxA + Bb * KT * HW;

    dcn_prep<<<(WB_WORDS + WOFF_WORDS + 255) / 256, 256, 0, stream>>>(w_dcn, w_off, wB, wOffB);
    dcn_offsets<<<Bb * Hh, 256, 0, stream>>>(x, wOffB, b_off, pyA, pxA, mA);
    dcn_main<<<Bb * Hh, 256, 0, stream>>>(x, wB, pyA, pxA, mA, out);
}
